// SplineGCN_88106959110528
// MI455X (gfx1250) — matrix operand order repaired and checked by reading, not compiled, not run on hardware
//
#include <hip/hip_runtime.h>
#include <math.h>

// ---------------------------------------------------------------------------
// SplineGCN (K=2, dim=1, degree=1, mean aggr) for gfx1250 / MI455X.
// Structure: GEMM (WMMA f16->f32) to produce [xa0|xa1|xroot], then edge
// gather/mix/atomic-scatter (L2-resident tables), then fused node epilogue.
// ---------------------------------------------------------------------------

typedef _Float16 f16;
typedef _Float16 v16h __attribute__((ext_vector_type(16)));
typedef float    v8f  __attribute__((ext_vector_type(8)));

#define F_IN   128
#define HIDDEN 64
#define NCLASS 40
#define CDIV(a, b) (((a) + (b) - 1) / (b))

union HalfFrag { v16h v; uint4 u[2]; };

// ---------------- small utility kernels ----------------

__global__ void k_fill_zero(float* __restrict__ p, int n) {
  int i = blockIdx.x * blockDim.x + threadIdx.x;
  if (i < n) p[i] = 0.0f;
}

__global__ void k_cvt_f16(const float* __restrict__ s, f16* __restrict__ d, int n) {
  int i = blockIdx.x * blockDim.x + threadIdx.x;
  if (i < n) d[i] = (f16)s[i];
}

// Pack [W1[0] | W1[1] | root1] as column-major f16: Bp[col*128 + k], 192 cols.
__global__ void k_pack_w1(const float* __restrict__ W1, const float* __restrict__ r1,
                          f16* __restrict__ Bp) {
  int i = blockIdx.x * blockDim.x + threadIdx.x;
  if (i >= 192 * 128) return;
  int col = i >> 7, k = i & 127;
  float v;
  if (col < 64)        v = W1[k * 64 + col];                // bank 0
  else if (col < 128)  v = W1[8192 + k * 64 + (col - 64)];  // bank 1
  else                 v = r1[k * 64 + (col - 128)];        // root
  Bp[col * 128 + k] = (f16)v;
}

// Pack [W2[0] | W2[1] | root2 | zero-pad] as column-major f16: Bp[col*64 + k], 128 cols.
__global__ void k_pack_w2(const float* __restrict__ W2, const float* __restrict__ r2,
                          f16* __restrict__ Bp) {
  int i = blockIdx.x * blockDim.x + threadIdx.x;
  if (i >= 128 * 64) return;
  int col = i >> 6, k = i & 63;
  float v = 0.0f;
  if (col < 40)        v = W2[k * 40 + col];                // bank 0
  else if (col < 80)   v = W2[2560 + k * 40 + (col - 40)];  // bank 1
  else if (col < 120)  v = r2[k * 40 + (col - 80)];         // root
  Bp[col * 64 + k] = (f16)v;
}

// ---------------- WMMA GEMM: C[M, LDC] = A[M, KDIM] * Bp (col-major) ----------------
// One wave per 16-row strip; A fragments held in registers across NT col tiles.
// A frag (16x32 f16): lane holds row (lane&15); halves p: K = 32*ks + 8*(lane>>4)
//   + 16*(p>>3) + (p&7)  -> two contiguous uint4 loads at k0 and k0+16.
// B frag (32x16 f16): lane holds col (lane&15); halves p: K = 32*ks + 8*(lane>>4)
//   + 16*(p>>3) + (p&7) -> two uint4 loads at k0 and k0+16.
// C/D (16x16 f32): acc[r] is element (rowBase + r + 8*(lane>>4), ct*16 + (lane&15)).
// LDC is a template parameter so row-store offsets (r*LDC*4 B) become immediate
// ioffsets on global_store_b32; full strips take a branch-free store path.
template <int KDIM, int NT, int LDC>
__global__ void k_gemm_wmma(const f16* __restrict__ A, const f16* __restrict__ Bp,
                            float* __restrict__ C, int M) {
  const int wave = threadIdx.x >> 5;
  const int lane = threadIdx.x & 31;
  const int strip = blockIdx.x * (blockDim.x >> 5) + wave;
  const int rowBase = strip * 16;
  if (rowBase >= M) return;                 // wave-uniform: EXEC stays all-ones
  const int lm = lane & 15, lh = lane >> 4;
  constexpr int KS = KDIM / 32;

  HalfFrag a[KS];
  {
    int row = rowBase + lm;
    if (row >= M) row = M - 1;              // clamp loads; stores are guarded
    const f16* ap = A + (size_t)row * KDIM + (lh << 3);
#pragma unroll
    for (int ks = 0; ks < KS; ++ks) {
      a[ks].u[0] = *(const uint4*)(ap + ks * 32);
      a[ks].u[1] = *(const uint4*)(ap + ks * 32 + 16);
    }
  }

  const bool fullStrip = (rowBase + 16 <= M);
  // One 32-bit base index per wave; all row/tile offsets are immediates.
  float* cbase = C + (unsigned)((rowBase + (lh << 3)) * LDC + lm);

#pragma unroll
  for (int ct = 0; ct < NT; ++ct) {
    v8f acc = {};
    const f16* bp = Bp + (size_t)(ct * 16 + lm) * KDIM + (lh << 3);
#pragma unroll
    for (int ks = 0; ks < KS; ++ks) {
      HalfFrag b;
      b.u[0] = *(const uint4*)(bp + ks * 32);
      b.u[1] = *(const uint4*)(bp + ks * 32 + 16);
      acc = __builtin_amdgcn_wmma_f32_16x16x32_f16(false, a[ks].v, false, b.v,
                                                   (short)0, acc, false, false);
    }
    float* cq = cbase + ct * 16;
    if (fullStrip) {
#pragma unroll
      for (int r = 0; r < 8; ++r) cq[r * LDC] = acc[r];   // immediate offsets
    } else {
#pragma unroll
      for (int r = 0; r < 8; ++r) {
        if (rowBase + r + (lh << 3) < M) cq[r * LDC] = acc[r];
      }
    }
  }
}

// ---------------- edge phase: m = (1-w)*xa0[src] + w*xa1[src], scatter to tgt ----------
// 16 lanes per edge, 4 features per lane (float4 gathers, f32 atomics). Gather
// table (xa/ya) is L2-resident (77MB / 51MB < 192MB L2).
__global__ void k_edge(const int* __restrict__ ei, const float* __restrict__ ew,
                       const float* __restrict__ xa, float* __restrict__ agg,
                       float* __restrict__ deg, int E, int F, int ld, int off1) {
  int t = blockIdx.x * blockDim.x + threadIdx.x;
  int e = t >> 4, g = t & 15;
  if (e >= E) return;
  int s = ei[e];          // edge_index[0][e] = src
  int d = ei[E + e];      // edge_index[1][e] = tgt
  float w = ew[e];
  if (deg != nullptr && g == 0) atomicAdd(&deg[d], 1.0f);
  int j = g << 2;
  if (j >= F) return;
  const float4 a0 = *(const float4*)(xa + (size_t)s * ld + j);
  const float4 a1 = *(const float4*)(xa + (size_t)s * ld + off1 + j);
  float u = 1.0f - w;
  float m0 = u * a0.x + w * a1.x;
  float m1 = u * a0.y + w * a1.y;
  float m2 = u * a0.z + w * a1.z;
  float m3 = u * a0.w + w * a1.w;
  float* dst = agg + (size_t)d * F + j;
  atomicAdd(dst + 0, m0);
  atomicAdd(dst + 1, m1);
  atomicAdd(dst + 2, m2);
  atomicAdd(dst + 3, m3);
}

// ---------------- layer-1 epilogue: h = elu(agg/deg + xroot + b1) -> f16 ----------------
__global__ void k_node1(const float* __restrict__ agg, const float* __restrict__ deg,
                        const float* __restrict__ xa, const float* __restrict__ b1,
                        f16* __restrict__ hh, int N) {
  int i = blockIdx.x * blockDim.x + threadIdx.x;
  if (i >= N * HIDDEN) return;
  int n = i >> 6, j = i & 63;
  float dv = fmaxf(deg[n], 1.0f);
  float v = agg[i] / dv + xa[(size_t)n * 192 + 128 + j] + b1[j];
  v = v > 0.0f ? v : (expf(v) - 1.0f);     // elu
  hh[i] = (f16)v;
}

// ---------------- layer-2 epilogue + log_softmax, one wave32 per node -------------------
__global__ void k_node2(const float* __restrict__ agg, const float* __restrict__ deg,
                        const float* __restrict__ ya, const float* __restrict__ b2,
                        float* __restrict__ out, int N) {
  int wid = (blockIdx.x * blockDim.x + threadIdx.x) >> 5;
  int lane = threadIdx.x & 31;
  if (wid >= N) return;
  float dv = fmaxf(deg[wid], 1.0f);
  const float* ab = agg + (size_t)wid * NCLASS;
  const float* yb = ya + (size_t)wid * 128 + 80;   // root part of packed GEMM output
  float z0 = -1e30f, z1 = -1e30f;
  if (lane < NCLASS) z0 = ab[lane] / dv + yb[lane] + b2[lane];
  int l2 = lane + 32;
  if (l2 < NCLASS) z1 = ab[l2] / dv + yb[l2] + b2[l2];
  float m = fmaxf(z0, z1);
#pragma unroll
  for (int o = 16; o > 0; o >>= 1) m = fmaxf(m, __shfl_xor(m, o, 32));
  float s = 0.0f;
  if (lane < NCLASS) s += expf(z0 - m);
  if (l2 < NCLASS) s += expf(z1 - m);
#pragma unroll
  for (int o = 16; o > 0; o >>= 1) s += __shfl_xor(s, o, 32);
  float lse = m + logf(s);
  if (lane < NCLASS) out[(size_t)wid * NCLASS + lane] = z0 - lse;
  if (l2 < NCLASS) out[(size_t)wid * NCLASS + l2] = z1 - lse;
}

// ---------------------------------------------------------------------------

extern "C" void kernel_launch(void* const* d_in, const int* in_sizes, int n_in,
                              void* d_out, int out_size, void* d_ws, size_t ws_size,
                              hipStream_t stream) {
  const float* x  = (const float*)d_in[0];
  const int*   ei = (const int*)d_in[1];
  const float* ew = (const float*)d_in[2];
  const float* W1 = (const float*)d_in[3];
  const float* r1 = (const float*)d_in[4];
  const float* b1 = (const float*)d_in[5];
  const float* W2 = (const float*)d_in[6];
  const float* r2 = (const float*)d_in[7];
  const float* b2 = (const float*)d_in[8];
  float* out = (float*)d_out;

  const int N = in_sizes[0] / F_IN;   // 100000
  const int E = in_sizes[1] / 2;      // 1600000

  // Workspace carve (xa buffer reused as ya; agg reused for both layers).
  char* ws = (char*)d_ws;
  size_t off = 0;
  auto carve = [&](size_t bytes) -> char* {
    char* p = ws + off;
    off += (bytes + 255) & ~(size_t)255;
    return p;
  };
  f16*   xh  = (f16*)  carve((size_t)N * F_IN * sizeof(f16));      // x in f16
  f16*   Bp1 = (f16*)  carve((size_t)192 * 128 * sizeof(f16));     // packed layer-1 weights
  f16*   Bp2 = (f16*)  carve((size_t)128 * 64 * sizeof(f16));      // packed layer-2 weights
  float* xa  = (float*)carve((size_t)N * 192 * sizeof(float));     // [xa0|xa1|xr]; reused as ya[N,128]
  float* agg = (float*)carve((size_t)N * HIDDEN * sizeof(float));  // reused as agg2[N,40]
  float* deg = (float*)carve((size_t)N * sizeof(float));
  f16*   hh  = (f16*)  carve((size_t)N * HIDDEN * sizeof(f16));    // hidden activations, f16

  const int strips = CDIV(N, 16);
  const int gemmBlocks = CDIV(strips, 4);   // 4 waves (16-row strips) per block

  // ---- prep ----
  k_cvt_f16<<<CDIV(N * F_IN, 256), 256, 0, stream>>>(x, xh, N * F_IN);
  k_pack_w1<<<CDIV(192 * 128, 256), 256, 0, stream>>>(W1, r1, Bp1);
  k_pack_w2<<<CDIV(128 * 64, 256), 256, 0, stream>>>(W2, r2, Bp2);

  // ---- layer 1 ----
  k_gemm_wmma<128, 12, 192><<<gemmBlocks, 128, 0, stream>>>(xh, Bp1, xa, N);
  k_fill_zero<<<CDIV(N * HIDDEN, 256), 256, 0, stream>>>(agg, N * HIDDEN);
  k_fill_zero<<<CDIV(N, 256), 256, 0, stream>>>(deg, N);
  k_edge<<<CDIV(E * 16, 256), 256, 0, stream>>>(ei, ew, xa, agg, deg, E, HIDDEN, 192, 64);
  k_node1<<<CDIV(N * HIDDEN, 256), 256, 0, stream>>>(agg, deg, xa, b1, hh, N);

  // ---- layer 2 (ya overlays xa buffer) ----
  k_gemm_wmma<64, 8, 128><<<gemmBlocks, 128, 0, stream>>>(hh, Bp2, xa, N);
  k_fill_zero<<<CDIV(N * NCLASS, 256), 256, 0, stream>>>(agg, N * NCLASS);
  k_edge<<<CDIV(E * 16, 256), 256, 0, stream>>>(ei, ew, xa, agg, nullptr, E, NCLASS, 128, 40);
  k_node2<<<CDIV(N * 32, 256), 256, 0, stream>>>(agg, deg, xa, b2, out, N);
}
